// model4_42992622633179
// MI455X (gfx1250) — matrix operand order repaired and checked by reading, not compiled, not run on hardware
//
#include <hip/hip_runtime.h>
#include <hip/hip_bf16.h>
#include <math.h>

// ---------------- problem constants ----------------
#define HID   51
#define G4    204          // 4*HID
#define NT    13           // ceil(204/16) gate tiles
#define GP    208          // padded gate dim (NT*16)
#define TT    2048
#define BSZ   1024
#define BTILE 16           // batch rows per workgroup
#define WGSZ  128          // 4 waves of 32

// padded strides (halves) to avoid LDS bank conflicts:
// row/col stride = Kpad + 8 halves (+16B) -> consecutive rows shift 4 banks
#define KS0   72           // layer-0 K stride   (2+51 -> 64, +8 pad)
#define KS12  136          // layer-1/2 K stride (51+51 -> 128, +8 pad)
#define GR    212          // gate buffer row stride in floats (208 + 4)
#define H2S   52           // padded f32 row stride for h2 / head weights

typedef _Float16 v16h __attribute__((ext_vector_type(16)));
typedef _Float16 v8h  __attribute__((ext_vector_type(8)));
typedef float    v8f  __attribute__((ext_vector_type(8)));

// ---------------- workspace (d_ws) byte layout ----------------
// P0   : 208*72  f16  @ 0        (29952 B)
// P1   : 208*136 f16  @ 29952    (56576 B)
// P2   : 208*136 f16  @ 86528    (56576 B)
// bias : 3*208   f32  @ 143104   (2496 B)
// total copied verbatim to LDS: 145600 B = 9100 * 16B
#define WS_P0     0
#define WS_P1     29952
#define WS_P2     86528
#define WS_BIAS   143104
#define WS_COPYB  145600

// ---------------- pack kernel: fp32 weights -> f16 packed, K-concat, padded stride ----------------
__global__ void lstm_pack_kernel(
    const float* __restrict__ Wih0, const float* __restrict__ Whh0,
    const float* __restrict__ bih0, const float* __restrict__ bhh0,
    const float* __restrict__ Wih1, const float* __restrict__ Whh1,
    const float* __restrict__ bih1, const float* __restrict__ bhh1,
    const float* __restrict__ Wih2, const float* __restrict__ Whh2,
    const float* __restrict__ bih2, const float* __restrict__ bhh2,
    char* __restrict__ ws)
{
    _Float16* P0   = (_Float16*)(ws + WS_P0);
    _Float16* P1   = (_Float16*)(ws + WS_P1);
    _Float16* P2   = (_Float16*)(ws + WS_P2);
    float*    Bias = (float*)   (ws + WS_BIAS);

    const int tid    = blockIdx.x * blockDim.x + threadIdx.x;
    const int stride = gridDim.x * blockDim.x;

    // layer 0: [W_ih (k<2) | W_hh (2..52) | 0 pad]
    for (int i = tid; i < GP * KS0; i += stride) {
        int r = i / KS0, k = i - r * KS0;
        float v = 0.f;
        if (r < G4) {
            if (k < 2)       v = Wih0[r * 2 + k];
            else if (k < 53) v = Whh0[r * HID + (k - 2)];
        }
        P0[i] = (_Float16)v;
    }
    // layers 1,2: [W_ih (k<51) | W_hh (51..101) | 0 pad]
    for (int i = tid; i < GP * KS12; i += stride) {
        int r = i / KS12, k = i - r * KS12;
        float v1 = 0.f, v2 = 0.f;
        if (r < G4) {
            if (k < HID)          { v1 = Wih1[r * HID + k];          v2 = Wih2[r * HID + k]; }
            else if (k < 2 * HID) { v1 = Whh1[r * HID + (k - HID)];  v2 = Whh2[r * HID + (k - HID)]; }
        }
        P1[i] = (_Float16)v1;
        P2[i] = (_Float16)v2;
    }
    // combined biases, zero-padded to 208
    for (int i = tid; i < 3 * GP; i += stride) {
        int l = i / GP, r = i - l * GP;
        float v = 0.f;
        if (r < G4) {
            if (l == 0)      v = bih0[r] + bhh0[r];
            else if (l == 1) v = bih1[r] + bhh1[r];
            else             v = bih2[r] + bhh2[r];
        }
        Bias[i] = v;
    }
}

// ---------------- fast activations ----------------
// CDNA5 has a TRANS32 hardware tanh; use it if the builtin exists (guarded so
// compilation never breaks), else fall back to exp+rcp (overflow-safe).
#if __has_builtin(__builtin_amdgcn_tanhf)
#define HW_TANH(x) __builtin_amdgcn_tanhf(x)
#elif __has_builtin(__builtin_amdgcn_tanh_f32)
#define HW_TANH(x) __builtin_amdgcn_tanh_f32(x)
#endif

__device__ __forceinline__ float ftanh(float x) {
#ifdef HW_TANH
    return HW_TANH(x);
#else
    float ax = fabsf(x);
    float e  = __expf(-2.f * ax);                       // e in (0,1], no overflow
    float t  = (1.f - e) * __builtin_amdgcn_rcpf(1.f + e);
    return copysignf(t, x);
#endif
}
__device__ __forceinline__ float fsigm(float x) {
#ifdef HW_TANH
    return fmaf(0.5f, HW_TANH(0.5f * x), 0.5f);         // sigma(x) = 0.5 + 0.5*tanh(x/2)
#else
    return __builtin_amdgcn_rcpf(1.f + __expf(-x));
#endif
}

// ---------------- main kernel: one workgroup = 16 batch rows, full time loop ----------------
__global__ __launch_bounds__(WGSZ, 1)
void lstm_main_kernel(const float* __restrict__ input,
                      const float* __restrict__ timeidx,
                      const char*  __restrict__ ws,
                      const float* __restrict__ Wlin,
                      const float* __restrict__ blin,
                      float* __restrict__ out)
{
    extern __shared__ char smem[];
    _Float16* sW0   = (_Float16*)smem;                 //      0  208*72 halves
    _Float16* sW1   = sW0 + GP * KS0;                  //  29952  208*136
    _Float16* sW2   = sW1 + GP * KS12;                 //  86528  208*136
    float*    sbias = (float*)(sW2 + GP * KS12);       // 143104  3*208 f32
    _Float16* xh0   = (_Float16*)(sbias + 3 * GP);     // 145600  16*72
    _Float16* xh1   = xh0 + BTILE * KS0;               // 147904  16*136
    _Float16* xh2   = xh1 + BTILE * KS12;              // 152256  16*136
    float*    gbuf  = (float*)(xh2 + BTILE * KS12);    // 156608  16*212 f32
    float*    h2f   = gbuf + BTILE * GR;               // 170176  16*52 f32 (16B aligned)
    float*    sWlin = h2f + BTILE * H2S;               // 173504  2*52 f32 (padded rows)
    float*    sblin = sWlin + 2 * H2S;                 // 173920  2 f32
                                                       // total 173928 B

    const int tid   = threadIdx.x;
    const int lane  = tid & 31;
    const int wave  = tid >> 5;
    const int col   = lane & 15;      // A row M / B col N
    const int hi    = lane >> 4;      // K-subchunk selector within wave
    const int bbase = blockIdx.x * BTILE;

    // ---- init: bulk-copy packed weights+biases ws -> LDS ----
    {
        const uint4* gsrc = (const uint4*)ws;
        uint4*       ldst = (uint4*)smem;
        for (int i = tid; i < WS_COPYB / 16; i += WGSZ) ldst[i] = gsrc[i];
    }
    // zero activation buffers (xh0..xh2 contiguous, includes pad cols) and h2f
    for (int i = tid; i < BTILE * (KS0 + 2 * KS12); i += WGSZ) xh0[i] = (_Float16)0.f;
    for (int i = tid; i < BTILE * H2S; i += WGSZ)              h2f[i] = 0.f;
    // head weights at padded stride, pad element zeroed
    for (int i = tid; i < 2 * H2S; i += WGSZ) {
        int o = i / H2S, j = i - o * H2S;
        sWlin[i] = (j < HID) ? Wlin[o * HID + j] : 0.f;
    }
    if (tid < 2) sblin[tid] = blin[tid];
    // stage x(t=0)
    if (tid < BTILE) {
        xh0[tid * KS0 + 0] = (_Float16)input[(bbase + tid) * TT + 0];
        xh0[tid * KS0 + 1] = (_Float16)timeidx[(bbase + tid) * TT + 0];
    }
    __syncthreads();

    // ---- hoist time-invariant biases into registers (per-lane, per tile, per layer) ----
    float breg[3][4];
#pragma unroll
    for (int l = 0; l < 3; ++l)
#pragma unroll
        for (int ti = 0; ti < 4; ++ti) {
            int n = wave + 4 * ti;
            breg[l][ti] = (n < NT) ? sbias[l * GP + n * 16 + col] : 0.f;
        }

    // ---- cell state lives in registers: thread<->(row, j-set) mapping is fixed ----
    float creg[3][7];
#pragma unroll
    for (int l = 0; l < 3; ++l)
#pragma unroll
        for (int s = 0; s < 7; ++s) creg[l][s] = 0.f;

    // ---- helpers ----
    // one 16x16 gate tile: bias-initialized WMMA chain + D store
    auto tile = [&](const _Float16* sW, const v16h (&a)[4], const float bv,
                    const int n, const int kchunks, const int kstride) {
        v8f acc = {bv, bv, bv, bv, bv, bv, bv, bv};
#pragma unroll
        for (int kk = 0; kk < 4; ++kk)
            if (kk < kchunks) {
                const _Float16* bp = sW + (n * 16 + col) * kstride + kk * 32 + hi * 8;
                v8h blo = *(const v8h*)bp;
                v8h bhi = *(const v8h*)(bp + 16);
                v16h b = __builtin_shufflevector(blo, bhi,
                        0, 1, 2, 3, 4, 5, 6, 7, 8, 9, 10, 11, 12, 13, 14, 15);
                acc = __builtin_amdgcn_wmma_f32_16x16x32_f16(
                        false, a[kk], false, b, (short)0, acc, false, false);
            }
        float* gp = gbuf + (hi * 8) * GR + n * 16 + col;
#pragma unroll
        for (int r = 0; r < 8; ++r) gp[r * GR] = acc[r];
    };

    auto do_mm = [&](const _Float16* xh, const _Float16* sW, const float (&br)[4],
                     const int kchunks, const int kstride) {
        // A fragments are shared by every gate tile: load once per layer
        v16h a[4];
#pragma unroll
        for (int kk = 0; kk < 4; ++kk)
            if (kk < kchunks) {
                const _Float16* ap = xh + col * kstride + kk * 32 + hi * 8;
                v8h alo = *(const v8h*)ap;
                v8h ahi = *(const v8h*)(ap + 16);
                a[kk] = __builtin_shufflevector(alo, ahi,
                        0, 1, 2, 3, 4, 5, 6, 7, 8, 9, 10, 11, 12, 13, 14, 15);
            }
        // tiles 0..2 are in-range for every wave (wave+8 <= 11 < 13): no guard,
        // straight-line code so the 3 independent acc chains interleave.
#pragma unroll
        for (int ti = 0; ti < 3; ++ti)
            tile(sW, a, br[ti], wave + 4 * ti, kchunks, kstride);
        // 4th tile exists only for wave 0: literal n = 12
        if (wave == 0)
            tile(sW, a, br[3], 12, kchunks, kstride);
    };

    // cell update: 8 threads per batch row; s<6 is guard-free (j<=47), tail l8<3
    auto do_elem = [&](const int l, float (&cl)[7],
                       _Float16* own, const int ownOff, const int ownStride,
                       _Float16* nxt, const int nxtStride) {
        const int row = tid >> 3;
        const int l8  = tid & 7;
        const float* gr = gbuf + row * GR;
        auto body = [&](int s, int j) {
            float gi = gr[j];
            float gf = gr[HID + j];
            float gg = gr[2 * HID + j];
            float go = gr[3 * HID + j];
            float cn = fsigm(gf) * cl[s] + fsigm(gi) * ftanh(gg);
            float hn = fsigm(go) * ftanh(cn);
            cl[s] = cn;
            _Float16 hh = (_Float16)hn;
            own[row * ownStride + ownOff + j] = hh;    // this layer, next step
            if (nxt) nxt[row * nxtStride + j] = hh;    // next layer, this step
            if (l == 2) h2f[row * H2S + j] = hn;       // f32 copy for the head
        };
#pragma unroll
        for (int s = 0; s < 6; ++s) body(s, l8 + 8 * s);
        if (l8 < 3) body(6, l8 + 48);
    };

    // ---- time loop (serial recurrence), 6 barriers per step ----
#pragma unroll 1
    for (int t = 0; t < TT; ++t) {
        // layer 0
        do_mm(xh0, sW0, breg[0], 2, KS0);
        __syncthreads();
        do_elem(0, creg[0], xh0, 2, KS0, xh1, KS12);
        if (tid < BTILE && t + 1 < TT) {      // stage x(t+1), same phase as xh0 h-write
            xh0[tid * KS0 + 0] = (_Float16)input[(bbase + tid) * TT + (t + 1)];
            xh0[tid * KS0 + 1] = (_Float16)timeidx[(bbase + tid) * TT + (t + 1)];
        }
        __syncthreads();

        // layer 1
        do_mm(xh1, sW1, breg[1], 4, KS12);
        __syncthreads();
        do_elem(1, creg[1], xh1, HID, KS12, xh2, KS12);
        __syncthreads();

        // layer 2
        do_mm(xh2, sW2, breg[2], 4, KS12);
        __syncthreads();
        do_elem(2, creg[2], xh2, HID, KS12, nullptr, 0);
        __syncthreads();

        // linear head + softplus on wave 3 (lightest-loaded wave); waves 0-2 may
        // run ahead into step t+1 layer-0 matmul (no hazard: >=3 barriers separate
        // these h2f reads from their next writes). 13 aligned float4 dot, no tail.
        if (tid >= 96) {
            int row = tid & 15;
            int o   = (tid >> 4) & 1;
            const float4* h4 = (const float4*)(h2f + row * H2S);
            const float4* w4 = (const float4*)(sWlin + o * H2S);
            float accv = sblin[o];
#pragma unroll
            for (int q = 0; q < 13; ++q) {
                float4 av = h4[q], bv = w4[q];
                accv += av.x * bv.x + av.y * bv.y + av.z * bv.z + av.w * bv.w;
            }
            float res = accv;
            if (o == 1) res = (accv > 30.f) ? accv : log1pf(__expf(accv));
            out[((size_t)(bbase + row) * TT + t) * 2 + o] = res;
        }
    }
}

// ---------------- launch ----------------
extern "C" void kernel_launch(void* const* d_in, const int* in_sizes, int n_in,
                              void* d_out, int out_size, void* d_ws, size_t ws_size,
                              hipStream_t stream) {
    (void)in_sizes; (void)n_in; (void)out_size; (void)ws_size;
    const float* input = (const float*)d_in[0];
    const float* timei = (const float*)d_in[1];
    const float* Wih0  = (const float*)d_in[2];
    const float* Whh0  = (const float*)d_in[3];
    const float* bih0  = (const float*)d_in[4];
    const float* bhh0  = (const float*)d_in[5];
    const float* Wih1  = (const float*)d_in[6];
    const float* Whh1  = (const float*)d_in[7];
    const float* bih1  = (const float*)d_in[8];
    const float* bhh1  = (const float*)d_in[9];
    const float* Wih2  = (const float*)d_in[10];
    const float* Whh2  = (const float*)d_in[11];
    const float* bih2  = (const float*)d_in[12];
    const float* bhh2  = (const float*)d_in[13];
    const float* Wlin  = (const float*)d_in[14];
    const float* blin  = (const float*)d_in[15];
    char* ws = (char*)d_ws;

    lstm_pack_kernel<<<64, 256, 0, stream>>>(Wih0, Whh0, bih0, bhh0,
                                             Wih1, Whh1, bih1, bhh1,
                                             Wih2, Whh2, bih2, bhh2, ws);

    const size_t ldsBytes = 173928;
    lstm_main_kernel<<<BSZ / BTILE, WGSZ, ldsBytes, stream>>>(
        input, timei, ws, Wlin, blin, (float*)d_out);
}
